// Attention_19335942767064
// MI455X (gfx1250) — matrix operand order repaired and checked by reading, not compiled, not run on hardware
//
#include <hip/hip_runtime.h>
#include <hip/hip_bf16.h>
#include <stdint.h>

// ---------------------------------------------------------------------------
// MHA block for MI455X (gfx1250, wave32, WMMA).
//   x[2,2048,1024] f32 -> qkv gemm -> flash attention (bf16 wmma, f32 softmax)
//   -> out projection -> f32 out.
// All matmuls use v_wmma_f32_16x16x32_bf16.  Identity bf16 tile staging uses
// the gfx1250 async global->LDS path (builtin confirmed present by probe).
// ---------------------------------------------------------------------------

#define EMBED 1024
#define SEQ   2048
#define BATCH 2
#define HEADS 16
#define DHEAD 64
#define NROWS (BATCH * SEQ)        // 4096
#define SCALE 0.125f               // 64^-0.5

typedef __attribute__((ext_vector_type(16))) __bf16 v16bf;
typedef __attribute__((ext_vector_type(8)))  float  v8f;
typedef __attribute__((ext_vector_type(4)))  unsigned int v4u;
typedef __attribute__((ext_vector_type(4)))  int v4i;

union Frag16 { v16bf h; v4u q[2]; };

// ---- gfx1250 async global->LDS path (guarded; falls back cleanly) ----------
#if defined(__has_builtin)
#  if __has_builtin(__builtin_amdgcn_global_load_async_to_lds_b128) && \
      __has_builtin(__builtin_amdgcn_s_wait_asynccnt)
#    define HAVE_ASYNC_LDS 1
#  endif
#endif
#ifndef HAVE_ASYNC_LDS
#  define HAVE_ASYNC_LDS 0
#endif

__device__ __forceinline__ void copy16_g2lds(unsigned short* lds_dst,
                                             const unsigned short* gsrc) {
#if HAVE_ASYNC_LDS
    __builtin_amdgcn_global_load_async_to_lds_b128(
        (__attribute__((address_space(1))) v4i*)gsrc,
        (__attribute__((address_space(3))) v4i*)lds_dst, 0, 0);
#else
    *(v4u*)lds_dst = *(const v4u*)gsrc;
#endif
}

__device__ __forceinline__ void async_copies_done() {
#if HAVE_ASYNC_LDS
    __builtin_amdgcn_s_wait_asynccnt(0);
#endif
}

__device__ __forceinline__ unsigned short f2bf(float f) {
    unsigned u = __float_as_uint(f);
    u += 0x7FFFu + ((u >> 16) & 1u);     // round-to-nearest-even
    return (unsigned short)(u >> 16);
}

__device__ __forceinline__ unsigned int f2bf_pk(float lo, float hi) {
    return (unsigned int)f2bf(lo) | ((unsigned int)f2bf(hi) << 16);
}

__device__ __forceinline__ v8f wmma_bf16(const Frag16& a, const Frag16& b, v8f c) {
    return __builtin_amdgcn_wmma_f32_16x16x32_bf16(
        false, a.h, false, b.h, (short)0, c, false, false);
}

// ---------------------------------------------------------------------------
// Kernel 1: QKV projection.  C[4096,3072] = X[4096,1024] @ Wqkv + b,
// scattered into bf16 Q/K/V buffers with layout [b*h, s, 64]; Q pre-scaled.
// Block: 256 threads (8 waves), tile 128(M) x 128(N), k-step 32.
// ---------------------------------------------------------------------------
__global__ __launch_bounds__(256) void qkv_gemm_kernel(
    const float* __restrict__ x, const float* __restrict__ w,
    const float* __restrict__ bias,
    unsigned short* __restrict__ qb, unsigned short* __restrict__ kb,
    unsigned short* __restrict__ vb)
{
    __shared__ alignas(16) unsigned short As[128 * 40];   // [row][k] stride 40
    __shared__ alignas(16) unsigned short Bs[128 * 40];   // [col][k] stride 40 (transposed)

    const int tid  = threadIdx.x;
    const int lane = tid & 31;
    const int wave = tid >> 5;
    const int wm   = wave & 3;          // 4 waves along M (32 rows each)
    const int wn   = wave >> 2;         // 2 waves along N (64 cols each)
    const int l15  = lane & 15;
    const int m0   = blockIdx.y * 128;
    const int n0   = blockIdx.x * 128;

    v8f acc[2][4];
    const v8f z = {0.f,0.f,0.f,0.f,0.f,0.f,0.f,0.f};
#pragma unroll
    for (int i = 0; i < 2; ++i)
#pragma unroll
        for (int j = 0; j < 4; ++j) acc[i][j] = z;

    for (int kt = 0; kt < EMBED; kt += 32) {
        __syncthreads();
        // prefetch next k-tile (speculative; dropped if OOB)
        {
            int g = tid;
            int row = g >> 3, kq = (g & 7) * 4;
            __builtin_prefetch(x + (size_t)(m0 + row) * EMBED + kt + 32 + kq, 0, 1);
            int kk = g >> 5, nq = (g & 31) * 4;
            __builtin_prefetch(w + (size_t)(kt + 32 + kk) * 3072 + n0 + nq, 0, 1);
        }
        // A tile: 128x32 f32 -> bf16 LDS (pack pairs into 8B stores)
#pragma unroll
        for (int j = 0; j < 4; ++j) {
            int g = tid + j * 256;                 // 1024 float4 groups
            int row = g >> 3, kq = (g & 7) * 4;
            const float4 f = *(const float4*)(x + (size_t)(m0 + row) * EMBED + kt + kq);
            uint2 pk;
            pk.x = f2bf_pk(f.x, f.y);
            pk.y = f2bf_pk(f.z, f.w);
            *(uint2*)&As[row * 40 + kq] = pk;
        }
        // B tile: 32x128 f32, store transposed [col][k]
#pragma unroll
        for (int j = 0; j < 4; ++j) {
            int g = tid + j * 256;
            int kk = g >> 5, nq = (g & 31) * 4;
            const float4 f = *(const float4*)(w + (size_t)(kt + kk) * 3072 + n0 + nq);
            Bs[(nq + 0) * 40 + kk] = f2bf(f.x);
            Bs[(nq + 1) * 40 + kk] = f2bf(f.y);
            Bs[(nq + 2) * 40 + kk] = f2bf(f.z);
            Bs[(nq + 3) * 40 + kk] = f2bf(f.w);
        }
        __syncthreads();

        Frag16 a[2], b[4];
        const int abase = (lane >= 16) ? 8 : 0;    // A: K chunks {0-7,16-23}/{8-15,24-31}
#pragma unroll
        for (int i = 0; i < 2; ++i) {
            const unsigned short* p = &As[(wm * 32 + i * 16 + l15) * 40 + abase];
            a[i].q[0] = *(const v4u*)p;
            a[i].q[1] = *(const v4u*)(p + 16);
        }
        const int kb2 = (lane >= 16) ? 8 : 0;      // B: K chunks {0-7,16-23}/{8-15,24-31}
#pragma unroll
        for (int j = 0; j < 4; ++j) {
            const unsigned short* p = &Bs[(wn * 64 + j * 16 + l15) * 40 + kb2];
            b[j].q[0] = *(const v4u*)p;
            b[j].q[1] = *(const v4u*)(p + 16);
        }
#pragma unroll
        for (int i = 0; i < 2; ++i)
#pragma unroll
            for (int j = 0; j < 4; ++j)
                acc[i][j] = wmma_bf16(a[i], b[j], acc[i][j]);
    }

    // Epilogue: bias, Q-scale, scatter into [bh, s, 64] bf16
    const int half = (lane >= 16) ? 8 : 0;
#pragma unroll
    for (int i = 0; i < 2; ++i) {
#pragma unroll
        for (int j = 0; j < 4; ++j) {
            const int colg  = n0 + wn * 64 + j * 16 + l15;
            const int which = colg >> 10;
            const int h     = (colg >> 6) & 15;
            const int d     = colg & 63;
            const float bv  = bias[colg];
            unsigned short* dst = (which == 0) ? qb : ((which == 1) ? kb : vb);
            const float sc = (which == 0) ? SCALE : 1.f;
#pragma unroll
            for (int r = 0; r < 8; ++r) {
                const int rowg = m0 + wm * 32 + i * 16 + half + r;
                const int bidx = rowg >> 11;             // /2048
                const int s    = rowg & 2047;
                const int bh   = bidx * HEADS + h;
                dst[(((size_t)bh * SEQ + s) << 6) + d] = f2bf((acc[i][j][r] + bv) * sc);
            }
        }
    }
}

// ---------------------------------------------------------------------------
// Kernel 2: flash attention.  One block per (bh, 64-row q tile); 4 waves,
// each wave owns 16 q rows.  Online softmax in the WMMA C layout.
// ---------------------------------------------------------------------------
__global__ __launch_bounds__(128) void attn_kernel(
    const unsigned short* __restrict__ qb, const unsigned short* __restrict__ kb,
    const unsigned short* __restrict__ vb, unsigned short* __restrict__ ob)
{
    __shared__ alignas(16) unsigned short Vt[64 * 72];      // V^T: [d][n] stride 72
    __shared__ alignas(16) unsigned short Ps[4 * 16 * 72];  // per-wave P strips

    const int tid  = threadIdx.x;
    const int lane = tid & 31;
    const int w    = tid >> 5;          // 4 waves
    const int l15  = lane & 15;
    const int half = (lane >= 16) ? 8 : 0;
    const int qt   = blockIdx.x;        // 0..31 (q tile)
    const int bh   = blockIdx.y;        // 0..31
    const size_t base = (size_t)bh * SEQ * DHEAD;
    const int q0   = qt * 64;

    // Q fragments for this wave's 16 rows, two 32-wide K steps over dh=64
    Frag16 qf[2];
    const int ab  = (lane >= 16) ? 8 : 0;
    const int kb2 = (lane >= 16) ? 8 : 0;
    {
        const unsigned short* qrow = qb + base + (size_t)(q0 + w * 16 + l15) * DHEAD;
#pragma unroll
        for (int ks = 0; ks < 2; ++ks) {
            qf[ks].q[0] = *(const v4u*)(qrow + ks * 32 + ab);
            qf[ks].q[1] = *(const v4u*)(qrow + ks * 32 + ab + 16);
        }
    }

    float mrow[8], lrow[8];
    v8f oacc[4];
    const v8f z = {0.f,0.f,0.f,0.f,0.f,0.f,0.f,0.f};
#pragma unroll
    for (int r = 0; r < 8; ++r) { mrow[r] = -1e30f; lrow[r] = 0.f; }
#pragma unroll
    for (int jt = 0; jt < 4; ++jt) oacc[jt] = z;

    for (int nb = 0; nb < SEQ; nb += 64) {
        __syncthreads();
        // prefetch next K/V tiles (speculative)
        __builtin_prefetch(kb + base + (size_t)(nb + 64 + (tid >> 1)) * DHEAD + (tid & 1) * 32, 0, 1);
        __builtin_prefetch(vb + base + (size_t)(nb + 64 + (tid >> 1)) * DHEAD + (tid & 1) * 32, 0, 1);
        // stage V tile transposed into LDS (bf16): Vt[d][n]
#pragma unroll
        for (int j = 0; j < 4; ++j) {
            int g = tid + j * 128;                    // 512 8-elem groups
            int row = g >> 3, dq = (g & 7) * 8;
            v4u t = *(const v4u*)(vb + base + (size_t)(nb + row) * DHEAD + dq);
            unsigned short e[8];
            e[0] = t.x & 0xffff; e[1] = t.x >> 16;
            e[2] = t.y & 0xffff; e[3] = t.y >> 16;
            e[4] = t.z & 0xffff; e[5] = t.z >> 16;
            e[6] = t.w & 0xffff; e[7] = t.w >> 16;
#pragma unroll
            for (int i = 0; i < 8; ++i) Vt[(dq + i) * 72 + row] = e[i];
        }
        __syncthreads();

        // S = Q @ K^T : B-frag columns are K rows (contiguous in memory)
        v8f sacc[4];
#pragma unroll
        for (int t = 0; t < 4; ++t) sacc[t] = z;
#pragma unroll
        for (int t = 0; t < 4; ++t) {
            const unsigned short* krow = kb + base + (size_t)(nb + t * 16 + l15) * DHEAD;
#pragma unroll
            for (int ks = 0; ks < 2; ++ks) {
                Frag16 kf;
                kf.q[0] = *(const v4u*)(krow + ks * 32 + kb2);
                kf.q[1] = *(const v4u*)(krow + ks * 32 + kb2 + 16);
                sacc[t] = wmma_bf16(qf[ks], kf, sacc[t]);
            }
        }

        // online softmax (row stats shared across the 16-lane half via shfl_xor)
        float fac[8];
#pragma unroll
        for (int r = 0; r < 8; ++r) {
            float vmax = fmaxf(fmaxf(sacc[0][r], sacc[1][r]),
                               fmaxf(sacc[2][r], sacc[3][r]));
            vmax = fmaxf(vmax, __shfl_xor(vmax, 1, 16));
            vmax = fmaxf(vmax, __shfl_xor(vmax, 2, 16));
            vmax = fmaxf(vmax, __shfl_xor(vmax, 4, 16));
            vmax = fmaxf(vmax, __shfl_xor(vmax, 8, 16));
            const float mnew = fmaxf(mrow[r], vmax);
            fac[r]  = __expf(mrow[r] - mnew);
            mrow[r] = mnew;
        }
#pragma unroll
        for (int t = 0; t < 4; ++t)
#pragma unroll
            for (int r = 0; r < 8; ++r)
                sacc[t][r] = __expf(sacc[t][r] - mrow[r]);
#pragma unroll
        for (int r = 0; r < 8; ++r) {
            float rs = sacc[0][r] + sacc[1][r] + sacc[2][r] + sacc[3][r];
            rs += __shfl_xor(rs, 1, 16);
            rs += __shfl_xor(rs, 2, 16);
            rs += __shfl_xor(rs, 4, 16);
            rs += __shfl_xor(rs, 8, 16);
            lrow[r] = lrow[r] * fac[r] + rs;
        }
#pragma unroll
        for (int jt = 0; jt < 4; ++jt)
#pragma unroll
            for (int r = 0; r < 8; ++r) oacc[jt][r] *= fac[r];

        // P: C-layout -> LDS -> A-layout (per-wave strip, wave-local ordering)
#pragma unroll
        for (int t = 0; t < 4; ++t)
#pragma unroll
            for (int r = 0; r < 8; ++r)
                Ps[(w * 16 + half + r) * 72 + t * 16 + l15] = f2bf(sacc[t][r]);

        Frag16 pf[2];
        const unsigned short* prow = &Ps[(w * 16 + l15) * 72];
#pragma unroll
        for (int ks = 0; ks < 2; ++ks) {
            pf[ks].q[0] = *(const v4u*)(prow + ks * 32 + ab);
            pf[ks].q[1] = *(const v4u*)(prow + ks * 32 + ab + 16);
        }
        // O += P @ V  (B-frag columns = rows of Vt)
#pragma unroll
        for (int jt = 0; jt < 4; ++jt) {
            const unsigned short* vrow = &Vt[(jt * 16 + l15) * 72];
#pragma unroll
            for (int ks = 0; ks < 2; ++ks) {
                Frag16 vf;
                vf.q[0] = *(const v4u*)(vrow + ks * 32 + kb2);
                vf.q[1] = *(const v4u*)(vrow + ks * 32 + kb2 + 16);
                oacc[jt] = wmma_bf16(pf[ks], vf, oacc[jt]);
            }
        }
    }

    // write O (bf16) in [b, s, h*64] layout for the out-proj GEMM
    const int bidx = bh >> 4, h = bh & 15;
#pragma unroll
    for (int jt = 0; jt < 4; ++jt) {
#pragma unroll
        for (int r = 0; r < 8; ++r) {
            const int s = q0 + w * 16 + half + r;
            const float val = oacc[jt][r] / lrow[r];
            ob[(size_t)(bidx * SEQ + s) * EMBED + h * 64 + jt * 16 + l15] = f2bf(val);
        }
    }
}

// ---------------------------------------------------------------------------
// Kernel 3: output projection.  out[4096,1024] = O_bf16 @ Wout + b_out (f32).
// A tile is an identity bf16 copy -> async global->LDS; the async copies are
// issued first so their latency overlaps the B-tile f32->bf16 conversion VALU
// work, then s_wait_asynccnt(0) + barrier.
// ---------------------------------------------------------------------------
__global__ __launch_bounds__(256) void out_gemm_kernel(
    const unsigned short* __restrict__ a, const float* __restrict__ w,
    const float* __restrict__ bias, float* __restrict__ out)
{
    __shared__ alignas(16) unsigned short As[128 * 40];
    __shared__ alignas(16) unsigned short Bs[128 * 40];

    const int tid  = threadIdx.x;
    const int lane = tid & 31;
    const int wave = tid >> 5;
    const int wm   = wave & 3;
    const int wn   = wave >> 2;
    const int l15  = lane & 15;
    const int m0   = blockIdx.y * 128;
    const int n0   = blockIdx.x * 128;

    v8f acc[2][4];
    const v8f z = {0.f,0.f,0.f,0.f,0.f,0.f,0.f,0.f};
#pragma unroll
    for (int i = 0; i < 2; ++i)
#pragma unroll
        for (int j = 0; j < 4; ++j) acc[i][j] = z;

    for (int kt = 0; kt < EMBED; kt += 32) {
        __syncthreads();
        // A tile (bf16 identity copy): async global->LDS b128
#pragma unroll
        for (int j = 0; j < 2; ++j) {
            int g = tid + j * 256;                 // 512 groups of 8 bf16
            int row = g >> 2, kq = (g & 3) * 8;
            copy16_g2lds(&As[row * 40 + kq],
                         a + (size_t)(m0 + row) * EMBED + kt + kq);
        }
        // prefetch next A k-tile while converting B
        {
            int row = tid >> 2, kq = (tid & 3) * 8;
            __builtin_prefetch(a + (size_t)(m0 + row) * EMBED + kt + 32 + kq, 0, 1);
        }
        // B tile f32 -> bf16, transposed (overlaps async A copies)
#pragma unroll
        for (int j = 0; j < 4; ++j) {
            int g = tid + j * 256;
            int kk = g >> 5, nq = (g & 31) * 4;
            const float4 f = *(const float4*)(w + (size_t)(kt + kk) * EMBED + n0 + nq);
            Bs[(nq + 0) * 40 + kk] = f2bf(f.x);
            Bs[(nq + 1) * 40 + kk] = f2bf(f.y);
            Bs[(nq + 2) * 40 + kk] = f2bf(f.z);
            Bs[(nq + 3) * 40 + kk] = f2bf(f.w);
        }
        async_copies_done();
        __syncthreads();

        Frag16 afr[2], bfr[4];
        const int abase = (lane >= 16) ? 8 : 0;
#pragma unroll
        for (int i = 0; i < 2; ++i) {
            const unsigned short* p = &As[(wm * 32 + i * 16 + l15) * 40 + abase];
            afr[i].q[0] = *(const v4u*)p;
            afr[i].q[1] = *(const v4u*)(p + 16);
        }
        const int kb2 = (lane >= 16) ? 8 : 0;
#pragma unroll
        for (int j = 0; j < 4; ++j) {
            const unsigned short* p = &Bs[(wn * 64 + j * 16 + l15) * 40 + kb2];
            bfr[j].q[0] = *(const v4u*)p;
            bfr[j].q[1] = *(const v4u*)(p + 16);
        }
#pragma unroll
        for (int i = 0; i < 2; ++i)
#pragma unroll
            for (int j = 0; j < 4; ++j)
                acc[i][j] = wmma_bf16(afr[i], bfr[j], acc[i][j]);
    }

    const int half = (lane >= 16) ? 8 : 0;
#pragma unroll
    for (int i = 0; i < 2; ++i) {
#pragma unroll
        for (int j = 0; j < 4; ++j) {
            const int colg = n0 + wn * 64 + j * 16 + l15;
            const float bv = bias[colg];
#pragma unroll
            for (int r = 0; r < 8; ++r) {
                const int rowg = m0 + wm * 32 + i * 16 + half + r;
                out[(size_t)rowg * EMBED + colg] = acc[i][j][r] + bv;
            }
        }
    }
}

// ---------------------------------------------------------------------------
extern "C" void kernel_launch(void* const* d_in, const int* in_sizes, int n_in,
                              void* d_out, int out_size, void* d_ws, size_t ws_size,
                              hipStream_t stream) {
    const float* x     = (const float*)d_in[0];
    const float* w_qkv = (const float*)d_in[1];
    const float* b_qkv = (const float*)d_in[2];
    const float* w_out = (const float*)d_in[3];
    const float* b_out = (const float*)d_in[4];
    float* out = (float*)d_out;

    const size_t HEAD_BYTES = (size_t)BATCH * HEADS * SEQ * DHEAD * sizeof(unsigned short); // 8 MB
    unsigned short* qb = (unsigned short*)d_ws;
    unsigned short* kb = (unsigned short*)((char*)d_ws + HEAD_BYTES);
    unsigned short* vb = (unsigned short*)((char*)d_ws + 2 * HEAD_BYTES);
    unsigned short* ob = (unsigned short*)((char*)d_ws + 3 * HEAD_BYTES);

    // 1) QKV projection into per-head bf16 buffers (Q pre-scaled)
    qkv_gemm_kernel<<<dim3(3 * EMBED / 128, NROWS / 128), 256, 0, stream>>>(
        x, w_qkv, b_qkv, qb, kb, vb);

    // 2) flash attention: (q-tiles, b*h)
    attn_kernel<<<dim3(SEQ / 64, BATCH * HEADS), 128, 0, stream>>>(qb, kb, vb, ob);

    // 3) output projection
    out_gemm_kernel<<<dim3(EMBED / 128, NROWS / 128), 256, 0, stream>>>(
        ob, w_out, b_out, out);
}
